// LightweightCrossAttention_3204045603415
// MI455X (gfx1250) — matrix operand order repaired and checked by reading, not compiled, not run on hardware
//
#include <hip/hip_runtime.h>

#define D_MODEL 1024
#define BATCH   4096
#define NMOD    8
#define ROWS    (NMOD * BATCH)      // 32768 GEMM rows
#define LDA_H   (D_MODEL + 8)       // LDS row stride in halves (2064B: bank-period-coprime)

typedef __attribute__((ext_vector_type(16))) __bf16 v16bf;
typedef __attribute__((ext_vector_type(8)))  __bf16 v8bf;
typedef __attribute__((ext_vector_type(4)))  __bf16 v4bf;
typedef __attribute__((ext_vector_type(8)))  float  v8f;
typedef __attribute__((ext_vector_type(4)))  float  v4f;

static __device__ __forceinline__ v8f wmma_bf16(v16bf a, v16bf b, v8f c) {
  // D = A(16x32 bf16) * B(32x16 bf16) + C(16x16 f32)
  return __builtin_amdgcn_wmma_f32_16x16x32_bf16(
      /*neg_a=*/false, a, /*neg_b=*/false, b,
      /*c_mod=*/(short)0, c, /*reuse_a=*/false, /*reuse_b=*/false);
}

// ---------------------------------------------------------------------------
// Pass 1: WvT_bf16[d][e] = (bf16) Wv[e][d]   (so the combine GEMM is K-major)
// ---------------------------------------------------------------------------
__global__ __launch_bounds__(256) void wv_transpose_kernel(const float* __restrict__ Wv,
                                                           __bf16* __restrict__ WvT) {
  __shared__ float tile[32][33];
  const int e0 = blockIdx.x * 32;
  const int d0 = blockIdx.y * 32;
  const int r = threadIdx.x >> 5;   // 0..7
  const int c = threadIdx.x & 31;   // 0..31
#pragma unroll
  for (int rr = 0; rr < 32; rr += 8)
    tile[r + rr][c] = Wv[(size_t)(e0 + r + rr) * D_MODEL + d0 + c];
  __syncthreads();
#pragma unroll
  for (int rr = 0; rr < 32; rr += 8) {
    const int d = d0 + r + rr;
    WvT[(size_t)d * D_MODEL + e0 + c] = (__bf16)tile[c][r + rr];
  }
}

// ---------------------------------------------------------------------------
// Pass 2: Wc[o][d] = sum_e Wo[o][e] * WvT[d][e]   (bf16 WMMA, f32 accum)
// One 16x16 output tile per wave; 8 waves / block; 512 blocks cover 64x64 tiles.
// ---------------------------------------------------------------------------
__global__ __launch_bounds__(256) void weight_combine_kernel(const float* __restrict__ Wo,
                                                             const __bf16* __restrict__ WvT,
                                                             __bf16* __restrict__ Wc) {
  const int lane = threadIdx.x & 31;
  const int wv   = threadIdx.x >> 5;
  const int tile = blockIdx.x * 8 + wv;      // 0..4095
  const int to = tile >> 6;                  // o-tile  0..63
  const int td = tile & 63;                  // d-tile  0..63
  const int l15 = lane & 15;
  const int sel = lane >> 4;

  const float*  aRow = Wo  + (size_t)(to * 16 + l15) * D_MODEL;   // A: Wo row, K=e
  const __bf16* bRow = WvT + (size_t)(td * 16 + l15) * D_MODEL + sel * 16;

  v8f acc = {};
  for (int kb = 0; kb < D_MODEL; kb += 32) {
    // A fragment (contiguous, matches B): lane sel=0 -> K{0..15}, sel=1 -> K{16..31}
    const v4f f0 = *(const v4f*)(aRow + kb + sel * 16);
    const v4f f1 = *(const v4f*)(aRow + kb + sel * 16 + 4);
    const v4f f2 = *(const v4f*)(aRow + kb + sel * 16 + 8);
    const v4f f3 = *(const v4f*)(aRow + kb + sel * 16 + 12);
    v16bf a = {(__bf16)f0.x, (__bf16)f0.y, (__bf16)f0.z, (__bf16)f0.w,
               (__bf16)f1.x, (__bf16)f1.y, (__bf16)f1.z, (__bf16)f1.w,
               (__bf16)f2.x, (__bf16)f2.y, (__bf16)f2.z, (__bf16)f2.w,
               (__bf16)f3.x, (__bf16)f3.y, (__bf16)f3.z, (__bf16)f3.w};
    v16bf b = *(const v16bf*)(bRow + kb);    // B lane n holds 16 contiguous K of column n
    acc = wmma_bf16(a, b, acc);
  }
  // C layout: lane(0-15)=N, VGPR j -> M=j ; lane(16-31) -> M=8+j
  const int d = td * 16 + l15;
#pragma unroll
  for (int j = 0; j < 8; ++j) {
    const int o = to * 16 + sel * 8 + j;
    Wc[(size_t)o * D_MODEL + d] = (__bf16)acc[j];
  }
}

// ---------------------------------------------------------------------------
// Main fused kernel:  out[row][n] = sum_d kv[row][d]*Wc[n][d] + bo[n] + query[row%B][n]
// Block = 64-row strip of kv staged once as bf16 in LDS (129KB, 2 blocks/WGP),
// 8 waves laid out 4(M) x 2(N); each wave: 16 rows x 64 cols per N-chunk.
// ---------------------------------------------------------------------------
__global__ __launch_bounds__(256) void fused_attn_gemm_kernel(const float* __restrict__ kv,
                                                              const float* __restrict__ query,
                                                              const float* __restrict__ bo,
                                                              const __bf16* __restrict__ Wc,
                                                              float* __restrict__ out) {
  __shared__ __bf16 ldsA[64 * LDA_H];

  const int tid = threadIdx.x;
  const int rowBase = blockIdx.x * 64;

  // ---- Stage: 64 x 1024 f32 -> bf16 in LDS (coalesced 4KB rows, streaming hint)
  for (int c = 0; c < 64; ++c) {
    const v4f* gp = (const v4f*)(kv + ((size_t)(rowBase + c) << 10)) + tid;
    const v4f f = __builtin_nontemporal_load(gp);
    v4bf p = {(__bf16)f.x, (__bf16)f.y, (__bf16)f.z, (__bf16)f.w};
    *(v4bf*)(ldsA + c * LDA_H + tid * 4) = p;
  }
  __syncthreads();

  const int lane = tid & 31;
  const int wv   = tid >> 5;
  const int wm   = wv >> 1;        // 0..3 -> M strip
  const int wn   = wv & 1;         // 0..1 -> N half
  const int l15  = lane & 15;
  const int sel  = lane >> 4;

  // A fragment base (LDS): row = wm*16 + l15; halves {kb+sel*16 .. +7} and {kb+sel*16+8 .. +7}
  const __bf16* aLo = ldsA + (size_t)(wm * 16 + l15) * LDA_H + sel * 16;
  const __bf16* aHi = aLo + 8;

  for (int nc = 0; nc < 8; ++nc) {
    const int nbase = nc * 128 + wn * 64;

    const __bf16* bp0 = Wc + (size_t)(nbase +  0 + l15) * D_MODEL + sel * 16;
    const __bf16* bp1 = Wc + (size_t)(nbase + 16 + l15) * D_MODEL + sel * 16;
    const __bf16* bp2 = Wc + (size_t)(nbase + 32 + l15) * D_MODEL + sel * 16;
    const __bf16* bp3 = Wc + (size_t)(nbase + 48 + l15) * D_MODEL + sel * 16;

    v8f acc0 = {}, acc1 = {}, acc2 = {}, acc3 = {};
#pragma unroll 2
    for (int kb = 0; kb < D_MODEL; kb += 32) {
      const v8bf lo = *(const v8bf*)(aLo + kb);
      const v8bf hi = *(const v8bf*)(aHi + kb);
      const v16bf a = __builtin_shufflevector(lo, hi, 0, 1, 2, 3, 4, 5, 6, 7,
                                              8, 9, 10, 11, 12, 13, 14, 15);
      acc0 = wmma_bf16(a, *(const v16bf*)(bp0 + kb), acc0);
      acc1 = wmma_bf16(a, *(const v16bf*)(bp1 + kb), acc1);
      acc2 = wmma_bf16(a, *(const v16bf*)(bp2 + kb), acc2);
      acc3 = wmma_bf16(a, *(const v16bf*)(bp3 + kb), acc3);
    }

    // ---- Epilogue: + bo[n] + query[row & 4095][n], streamed f32 store
    const int rw = rowBase + wm * 16 + sel * 8;      // first of 8 rows this lane writes
#pragma unroll
    for (int nt = 0; nt < 4; ++nt) {
      const int n = nbase + nt * 16 + l15;
      const float bias = bo[n];
      const v8f acc = (nt == 0) ? acc0 : (nt == 1) ? acc1 : (nt == 2) ? acc2 : acc3;
#pragma unroll
      for (int j = 0; j < 8; ++j) {
        const int row = rw + j;
        const int b   = row & (BATCH - 1);
        const float val = acc[j] + bias + query[((size_t)b << 10) + n];
        __builtin_nontemporal_store(val, out + ((size_t)row << 10) + n);
      }
    }
  }
}

// ---------------------------------------------------------------------------
extern "C" void kernel_launch(void* const* d_in, const int* in_sizes, int n_in,
                              void* d_out, int out_size, void* d_ws, size_t ws_size,
                              hipStream_t stream) {
  (void)in_sizes; (void)n_in; (void)out_size; (void)ws_size;
  const float* query = (const float*)d_in[0];
  const float* kv    = (const float*)d_in[1];
  // d_in[2]=Wq, d_in[3]=Wk are dead code (softmax over a size-1 axis -> weights==1)
  const float* Wv    = (const float*)d_in[4];
  const float* Wo    = (const float*)d_in[5];
  const float* bo    = (const float*)d_in[6];
  float* out = (float*)d_out;

  __bf16* wvT = (__bf16*)d_ws;                                          // 2 MB
  __bf16* wc  = (__bf16*)((char*)d_ws + (size_t)D_MODEL * D_MODEL * 2); // 2 MB

  wv_transpose_kernel<<<dim3(32, 32), 256, 0, stream>>>(Wv, wvT);
  weight_combine_kernel<<<512, 256, 0, stream>>>(Wo, wvT, wc);
  fused_attn_gemm_kernel<<<ROWS / 64, 256, 0, stream>>>(kv, query, bo, wc, out);
}
